// NoFoDifformer_FourierKAN_7670811591204
// MI455X (gfx1250) — compile-verified, matrix operand order checked by reading, not run on hardware
//
#include <hip/hip_runtime.h>
#include <math.h>

// ---------------------------------------------------------------------------
// CDNA5 (gfx1250) WMMA types
// ---------------------------------------------------------------------------
typedef __bf16 bhalf;
typedef __attribute__((ext_vector_type(16))) __bf16 v16bf;
typedef __attribute__((ext_vector_type(8)))  float  v8f;

static __device__ __forceinline__ bhalf f2b(float x) { return (bhalf)x; }

#define BM 128
#define BN 128
#define BK 32
#define LDK 36   // padded LDS row stride (elements): 72B rows spread 64 banks

union FragU {
    v16bf v;
    unsigned u[8];
};

// pack two f32 -> one dword of 2 bf16: force a single v_cvt_pk_bf16_f32
static __device__ __forceinline__ unsigned pack2(float lo, float hi) {
    unsigned r;
    asm("v_cvt_pk_bf16_f32 %0, %1, %2" : "=v"(r) : "v"(lo), "v"(hi));
    return r;
}

// Load a 16x32 bf16 fragment (A-layout per CDNA5 ISA 7.12.2) from an LDS tile
// stored row-major [rows][LDK].  Constant chunk offsets -> ds_load_2addr_b64.
static __device__ __forceinline__ v16bf lds_frag16(const bhalf* base, int row0) {
    const int lane = threadIdx.x & 31;
    const int half = lane >> 4;
    const bhalf* p = base + (row0 + (lane & 15)) * LDK + half * 8;
    FragU f;
    const uint2 q0 = *(const uint2*)(p + 0);   // K 0..3   (+8*half)
    const uint2 q1 = *(const uint2*)(p + 4);   // K 4..7
    const uint2 q2 = *(const uint2*)(p + 16);  // K 16..19
    const uint2 q3 = *(const uint2*)(p + 20);  // K 20..23
    f.u[0] = q0.x; f.u[1] = q0.y; f.u[2] = q1.x; f.u[3] = q1.y;
    f.u[4] = q2.x; f.u[5] = q2.y; f.u[6] = q3.x; f.u[7] = q3.y;
    return f.v;
}

// Transposed 4x4 micro-tile store: r[j] holds 4 consecutive "row" values of
// k-row (kq+j).  Packs (k,k+1) pairs and emits 4 ds_store_b64.
static __device__ __forceinline__ void store4x4(bhalf* S, const float4 (&r)[4],
                                                int row0, int kq) {
    uint2 q;
    q.x = pack2(r[0].x, r[1].x); q.y = pack2(r[2].x, r[3].x);
    *(uint2*)(S + (row0 + 0) * LDK + kq) = q;
    q.x = pack2(r[0].y, r[1].y); q.y = pack2(r[2].y, r[3].y);
    *(uint2*)(S + (row0 + 1) * LDK + kq) = q;
    q.x = pack2(r[0].z, r[1].z); q.y = pack2(r[2].z, r[3].z);
    *(uint2*)(S + (row0 + 2) * LDK + kq) = q;
    q.x = pack2(r[0].w, r[1].w); q.y = pack2(r[2].w, r[3].w);
    *(uint2*)(S + (row0 + 3) * LDK + kq) = q;
}

// ---------------------------------------------------------------------------
// Register-staged tile loaders.  Full tiles are branch-free: rows >= M are
// clamped (their products only reach output rows >= M, which the epilogue
// discards).  M is a multiple of 4, so float4 groups never straddle M.
// Non-TA A: thread -> (row r, 16 k).   TA A / B: thread -> (4 k-rows, 4 cols).
// ---------------------------------------------------------------------------
template <bool TA>
static __device__ __forceinline__ void load_a_full(const float* __restrict__ A,
                                                   int m0blk, int kbase, int M,
                                                   int lda, float4 (&ra)[4]) {
    const int t = threadIdx.x;
    if (!TA) {
        const int r = t >> 1, c0 = (t & 1) * 16;
        int grow = m0blk + r;
        if (grow >= M) grow = 0;  // don't-care row
        const float* ap = A + (size_t)grow * lda + kbase + c0;
#pragma unroll
        for (int j = 0; j < 4; ++j) ra[j] = *(const float4*)(ap + 4 * j);
    } else {
        const int kq = (t >> 5) * 4, m4 = (t & 31) * 4;
        int mg = m0blk + m4;
        if (mg >= M) mg = 0;  // don't-care rows
#pragma unroll
        for (int j = 0; j < 4; ++j)
            ra[j] = *(const float4*)(A + (size_t)(kbase + kq + j) * lda + mg);
    }
}

template <bool TA>
static __device__ __forceinline__ void load_a_tail(const float* __restrict__ A,
                                                   int m0blk, int kbase, int M,
                                                   int K, int lda, float4 (&ra)[4]) {
    const int t = threadIdx.x;
    const float4 z = make_float4(0.f, 0.f, 0.f, 0.f);
    if (!TA) {
        const int r = t >> 1, c0 = (t & 1) * 16;
        int grow = m0blk + r;
        if (grow >= M) grow = 0;
        const float* ap = A + (size_t)grow * lda + kbase + c0;
#pragma unroll
        for (int j = 0; j < 4; ++j) {
            const int kg = kbase + c0 + 4 * j;
            float4 f = z;
            if (kg + 3 < K) {
                f = *(const float4*)(ap + 4 * j);
            } else {
                if (kg + 0 < K) f.x = ap[4 * j + 0];
                if (kg + 1 < K) f.y = ap[4 * j + 1];
                if (kg + 2 < K) f.z = ap[4 * j + 2];
            }
            ra[j] = f;
        }
    } else {
        const int kq = (t >> 5) * 4, m4 = (t & 31) * 4;
        int mg = m0blk + m4;
        if (mg >= M) mg = 0;
#pragma unroll
        for (int j = 0; j < 4; ++j) {
            const int gk = kbase + kq + j;
            ra[j] = (gk < K) ? *(const float4*)(A + (size_t)gk * lda + mg) : z;
        }
    }
}

static __device__ __forceinline__ void load_b_full(const float* __restrict__ B,
                                                   int kbase, float4 (&rb)[4]) {
    const int t = threadIdx.x;
    const int kq = (t >> 5) * 4, n4 = (t & 31) * 4;
#pragma unroll
    for (int j = 0; j < 4; ++j)
        rb[j] = *(const float4*)(B + (size_t)(kbase + kq + j) * 128 + n4);
}

static __device__ __forceinline__ void load_b_tail(const float* __restrict__ B,
                                                   int kbase, int K, float4 (&rb)[4]) {
    const int t = threadIdx.x;
    const int kq = (t >> 5) * 4, n4 = (t & 31) * 4;
    const float4 z = make_float4(0.f, 0.f, 0.f, 0.f);
#pragma unroll
    for (int j = 0; j < 4; ++j) {
        const int gk = kbase + kq + j;
        rb[j] = (gk < K) ? *(const float4*)(B + (size_t)gk * 128 + n4) : z;
    }
}

// reg -> LDS (fp32 -> bf16 convert)
template <bool TA>
static __device__ __forceinline__ void store_a_lds(bhalf* As, const float4 (&ra)[4]) {
    const int t = threadIdx.x;
    if (!TA) {
        const int r = t >> 1, c0 = (t & 1) * 16;
        union { unsigned u[8]; uint2 q[4]; } p;
#pragma unroll
        for (int j = 0; j < 4; ++j) {
            p.u[2 * j + 0] = pack2(ra[j].x, ra[j].y);
            p.u[2 * j + 1] = pack2(ra[j].z, ra[j].w);
        }
        uint2* d = (uint2*)(As + r * LDK + c0);  // 8B aligned: 72*r + {0,32}
#pragma unroll
        for (int j = 0; j < 4; ++j) d[j] = p.q[j];
    } else {
        const int kq = (t >> 5) * 4, m4 = (t & 31) * 4;
        store4x4(As, ra, m4, kq);
    }
}

static __device__ __forceinline__ void store_b_lds(bhalf* Bs, const float4 (&rb)[4]) {
    const int t = threadIdx.x;
    const int kq = (t >> 5) * 4, n4 = (t & 31) * 4;
    store4x4(Bs, rb, n4, kq);
}

// ---------------------------------------------------------------------------
// Epilogue helper (GUARD=false for interior blocks -> no per-row branches)
// ---------------------------------------------------------------------------
template <int ACT, bool GUARD>
static __device__ __forceinline__ void do_epilogue(
    const v8f (&acc)[2][4], float alpha, const float* __restrict__ bias,
    const float* __restrict__ row_scale, const float* __restrict__ res0,
    const float* __restrict__ res1, int accum, float* __restrict__ C, int M,
    int m0blk, int wm, int wn, int lane) {
    const int half = lane >> 4;
    const int ln = lane & 15;
#pragma unroll
    for (int mt = 0; mt < 2; ++mt) {
#pragma unroll
        for (int nt = 0; nt < 4; ++nt) {
            const int col = wn + nt * 16 + ln;
            const float bv = bias ? bias[col] : 0.f;
#pragma unroll
            for (int r = 0; r < 8; ++r) {
                const int row = m0blk + wm + mt * 16 + half * 8 + r;
                if (GUARD && row >= M) continue;
                float v = alpha * acc[mt][nt][r] + bv;
                if (ACT == 1) v = fmaxf(v, 0.f);
                else if (ACT == 2) v = 0.5f * v * (1.f + erff(v * 0.70710678118654752f));
                if (row_scale) v *= row_scale[row];
                const size_t idx = (size_t)row * 128 + col;
                if (accum) v += C[idx];
                if (res0) v += res0[idx];
                if (res1) v += res1[idx];
                C[idx] = v;
            }
        }
    }
}

// ---------------------------------------------------------------------------
// Generic fused WMMA GEMM (double-buffered LDS, register-staged prefetch):
//   C = epi( alpha * op(A) @ B + bias ), B is [K,128], C is [M,128].
// ---------------------------------------------------------------------------
template <int ACT, bool TA>
__global__ __launch_bounds__(256)
void gemm_k(const float* __restrict__ A, const float* __restrict__ B,
            const float* __restrict__ bias, const float* __restrict__ alpha_ptr,
            float alpha_c, const float* __restrict__ row_scale,
            const float* __restrict__ res0, const float* __restrict__ res1,
            int accum, float* __restrict__ C, int M, int K, int lda) {
    __shared__ __align__(16) bhalf As[2][BM * LDK];
    __shared__ __align__(16) bhalf Bs[2][BN * LDK];

    const int t = threadIdx.x;
    const int m0blk = blockIdx.x * BM;
    const int lane = t & 31;
    const int w = t >> 5;
    const int wm = (w >> 1) * 32;  // 4 row-waves
    const int wn = (w & 1) * 64;   // 2 col-waves

    v8f acc[2][4];
    const v8f vzero = {0.f, 0.f, 0.f, 0.f, 0.f, 0.f, 0.f, 0.f};
#pragma unroll
    for (int i = 0; i < 2; ++i)
#pragma unroll
        for (int j = 0; j < 4; ++j) acc[i][j] = vzero;

    const int nk = (K + BK - 1) / BK;
    const int nk_full = (K % BK) ? nk - 1 : nk;  // K >= 128 so nk_full >= 1

    float4 ra[4], rb[4];
    load_a_full<TA>(A, m0blk, 0, M, lda, ra);
    load_b_full(B, 0, rb);
    store_a_lds<TA>(As[0], ra);
    store_b_lds(Bs[0], rb);
    __syncthreads();

    for (int kb = 0; kb < nk; ++kb) {
        const int cur = kb & 1;
        const bool have_next = (kb + 1 < nk);
        if (have_next) {
            const int kn = (kb + 1) * BK;
            if (kb + 1 < nk_full) {
                load_a_full<TA>(A, m0blk, kn, M, lda, ra);
                load_b_full(B, kn, rb);
            } else {
                load_a_tail<TA>(A, m0blk, kn, M, K, lda, ra);
                load_b_tail(B, kn, K, rb);
            }
        }
        // compute current tile while next-tile global loads are in flight
        v16bf afr[2], bfr[4];
        const bhalf* Ac = As[cur];
        const bhalf* Bc = Bs[cur];
        afr[0] = lds_frag16(Ac, wm);
        afr[1] = lds_frag16(Ac, wm + 16);
#pragma unroll
        for (int nt = 0; nt < 4; ++nt) bfr[nt] = lds_frag16(Bc, wn + nt * 16);
#pragma unroll
        for (int mt = 0; mt < 2; ++mt)
#pragma unroll
            for (int nt = 0; nt < 4; ++nt)
                acc[mt][nt] = __builtin_amdgcn_wmma_f32_16x16x32_bf16(
                    false, afr[mt], false, bfr[nt], (short)0, acc[mt][nt], false, false);
        if (have_next) {
            store_a_lds<TA>(As[cur ^ 1], ra);
            store_b_lds(Bs[cur ^ 1], rb);
            __syncthreads();
        }
    }

    float alpha = alpha_c;
    if (alpha_ptr) alpha *= *alpha_ptr;
    if (m0blk + BM <= M)
        do_epilogue<ACT, false>(acc, alpha, bias, row_scale, res0, res1, accum, C, M,
                                m0blk, wm, wn, lane);
    else
        do_epilogue<ACT, true>(acc, alpha, bias, row_scale, res0, res1, accum, C, M,
                               m0blk, wm, wn, lane);
}

// ---------------------------------------------------------------------------
// FourierKAN spectral filter
// ---------------------------------------------------------------------------
__global__ void k_fourier(const float* __restrict__ e, const float* __restrict__ ka,
                          const float* __restrict__ kb, const float* __restrict__ kbias,
                          const float* __restrict__ alpha_w, float* __restrict__ new_e,
                          int n) {
    const int i = blockIdx.x * blockDim.x + threadIdx.x;
    if (i >= n) return;
    const float t = e[i] * 0.31830988618379067f;  // e / pi
    float phi = kbias[0];
#pragma unroll
    for (int k = 1; k <= 10; ++k) {
        const float ang = t * (float)k;
        phi += ka[k - 1] * __cosf(ang) + kb[k - 1] * __sinf(ang);
    }
    new_e[i] = phi * alpha_w[0];
}

// ---------------------------------------------------------------------------
// LayerNorm over 128 cols, one wave per row:  y = (ln(x)*g + b) * outscale
// ---------------------------------------------------------------------------
__global__ void k_layernorm(const float* __restrict__ x, const float* __restrict__ g,
                            const float* __restrict__ b, float* __restrict__ y,
                            int M, float outscale) {
    const int lane = threadIdx.x & 31;
    const int row = blockIdx.x * 8 + (threadIdx.x >> 5);
    if (row >= M) return;
    const float* xr = x + (size_t)row * 128;
    float4 v = *(const float4*)(xr + lane * 4);
    float s = v.x + v.y + v.z + v.w;
#pragma unroll
    for (int off = 16; off > 0; off >>= 1) s += __shfl_xor(s, off);
    const float mean = s * (1.f / 128.f);
    const float d0 = v.x - mean, d1 = v.y - mean, d2 = v.z - mean, d3 = v.w - mean;
    float q = d0 * d0 + d1 * d1 + d2 * d2 + d3 * d3;
#pragma unroll
    for (int off = 16; off > 0; off >>= 1) q += __shfl_xor(q, off);
    const float rstd = rsqrtf(q * (1.f / 128.f) + 1e-5f);
    const int c = lane * 4;
    float4 o;
    o.x = (d0 * rstd * g[c + 0] + b[c + 0]) * outscale;
    o.y = (d1 * rstd * g[c + 1] + b[c + 1]) * outscale;
    o.z = (d2 * rstd * g[c + 2] + b[c + 2]) * outscale;
    o.w = (d3 * rstd * g[c + 3] + b[c + 3]) * outscale;
    *(float4*)(y + (size_t)row * 128 + c) = o;
}

// lam_full = exp(sum lq1*lk1) - exp(sum lq2*lk2) + LAMBDA_INIT(0.2)
__global__ void k_lambda(const float* __restrict__ lq1, const float* __restrict__ lk1,
                         const float* __restrict__ lq2, const float* __restrict__ lk2,
                         float* __restrict__ lam) {
    const int lane = threadIdx.x;
    float s1 = 0.f, s2 = 0.f;
    for (int j = lane; j < 128; j += 32) {
        s1 += lq1[j] * lk1[j];
        s2 += lq2[j] * lk2[j];
    }
#pragma unroll
    for (int off = 16; off > 0; off >>= 1) {
        s1 += __shfl_xor(s1, off);
        s2 += __shfl_xor(s2, off);
    }
    if (lane == 0) *lam = expf(s1) - expf(s2) + 0.2f;
}

// ---------------------------------------------------------------------------
extern "C" void kernel_launch(void* const* d_in, const int* in_sizes, int n_in,
                              void* d_out, int out_size, void* d_ws, size_t ws_size,
                              hipStream_t stream) {
    (void)in_sizes; (void)n_in; (void)out_size; (void)ws_size;
    const int N = 10000, NFEAT = 512, NC = 128;

    const float* e_in   = (const float*)d_in[0];
    const float* u      = (const float*)d_in[1];
    const float* x      = (const float*)d_in[2];
    const float* fe_w1  = (const float*)d_in[3];
    const float* fe_b1  = (const float*)d_in[4];
    const float* fe_w2  = (const float*)d_in[5];
    const float* fe_b2  = (const float*)d_in[6];
    const float* kan_a  = (const float*)d_in[7];
    const float* kan_b  = (const float*)d_in[8];
    const float* kan_bs = (const float*)d_in[9];
    const float* alpha_w= (const float*)d_in[10];
    const float* mha_g  = (const float*)d_in[11];
    const float* mha_b  = (const float*)d_in[12];
    const float* ffn_g  = (const float*)d_in[13];
    const float* ffn_b  = (const float*)d_in[14];
    const float* q1_w = (const float*)d_in[15]; const float* q1_b = (const float*)d_in[16];
    const float* k1_w = (const float*)d_in[17]; const float* k1_b = (const float*)d_in[18];
    const float* q2_w = (const float*)d_in[19]; const float* q2_b = (const float*)d_in[20];
    const float* k2_w = (const float*)d_in[21]; const float* k2_b = (const float*)d_in[22];
    const float* v_w  = (const float*)d_in[23]; const float* v_b  = (const float*)d_in[24];
    const float* att_g= (const float*)d_in[25]; const float* att_b= (const float*)d_in[26];
    const float* out_w= (const float*)d_in[27]; const float* out_b= (const float*)d_in[28];
    const float* lq1 = (const float*)d_in[29]; const float* lk1 = (const float*)d_in[30];
    const float* lq2 = (const float*)d_in[31]; const float* lk2 = (const float*)d_in[32];
    const float* f1_w= (const float*)d_in[33]; const float* f1_b= (const float*)d_in[34];
    const float* f2_w= (const float*)d_in[35]; const float* f2_b= (const float*)d_in[36];
    float* out = (float*)d_out;

    // workspace layout (floats)
    float* ws = (float*)d_ws;
    float* new_e = ws;                       // 10000
    float* lam   = ws + 10240;               // 1
    float* k1v   = ws + 10304;               // 128*128
    float* k2v   = ws + 10304 + 16384;       // 128*128
    const size_t SZ = (size_t)N * NC;
    float* base = ws + 65536;
    float* h1   = base + 0 * SZ;             // fe hidden, later gelu buf
    float* h    = base + 1 * SZ;
    float* hn   = base + 2 * SZ;
    float* q1   = base + 3 * SZ;
    float* k1   = base + 4 * SZ;
    float* q2   = base + 5 * SZ;
    float* k2   = base + 6 * SZ;
    float* vv   = base + 7 * SZ;
    float* s    = base + 8 * SZ;
    float* henc = base + 9 * SZ;
    float* apre = base + 10 * SZ;
    float* mh   = base + 11 * SZ;
    float* f    = base + 12 * SZ;
    float* g1   = h1;

    const dim3 blk(256);
    const int GB = (N + BM - 1) / BM;        // 79 row blocks
    const int LNG = (N + 7) / 8;

    // spectral filter coefficients
    k_fourier<<<(N + 255) / 256, 256, 0, stream>>>(e_in, kan_a, kan_b, kan_bs, alpha_w, new_e, N);

    // feat encoder: h = relu(x @ fe_w1 + b1) @ fe_w2 + b2
    gemm_k<1, false><<<GB, blk, 0, stream>>>(x,  fe_w1, fe_b1, nullptr, 1.f, nullptr, nullptr, nullptr, 0, h1, N, NFEAT, NFEAT);
    gemm_k<0, false><<<GB, blk, 0, stream>>>(h1, fe_w2, fe_b2, nullptr, 1.f, nullptr, nullptr, nullptr, 0, h,  N, NC, NC);

    // hn = LN(h)
    k_layernorm<<<LNG, 256, 0, stream>>>(h, mha_g, mha_b, hn, N, 1.0f);

    // spectral propagation: s = new_e (.) (u^T @ h);  h_enc = u @ s
    gemm_k<0, true ><<<GB, blk, 0, stream>>>(u, h, nullptr, nullptr, 1.f, new_e,  nullptr, nullptr, 0, s,    N, N, N);
    gemm_k<0, false><<<GB, blk, 0, stream>>>(u, s, nullptr, nullptr, 1.f, nullptr, nullptr, nullptr, 0, henc, N, N, N);

    // projections
    gemm_k<0, false><<<GB, blk, 0, stream>>>(hn, q1_w, q1_b, nullptr, 1.f, nullptr, nullptr, nullptr, 0, q1, N, NC, NC);
    gemm_k<0, false><<<GB, blk, 0, stream>>>(hn, k1_w, k1_b, nullptr, 1.f, nullptr, nullptr, nullptr, 0, k1, N, NC, NC);
    gemm_k<0, false><<<GB, blk, 0, stream>>>(hn, q2_w, q2_b, nullptr, 1.f, nullptr, nullptr, nullptr, 0, q2, N, NC, NC);
    gemm_k<0, false><<<GB, blk, 0, stream>>>(hn, k2_w, k2_b, nullptr, 1.f, nullptr, nullptr, nullptr, 0, k2, N, NC, NC);
    gemm_k<0, false><<<GB, blk, 0, stream>>>(hn, v_w,  v_b,  nullptr, 1.f, nullptr, nullptr, nullptr, 0, vv, N, NC, NC);

    // k1v = k1^T @ v, k2v = k2^T @ v   (single 128x128 block each)
    gemm_k<0, true ><<<1, blk, 0, stream>>>(k1, vv, nullptr, nullptr, 1.f, nullptr, nullptr, nullptr, 0, k1v, NC, N, NC);
    gemm_k<0, true ><<<1, blk, 0, stream>>>(k2, vv, nullptr, nullptr, 1.f, nullptr, nullptr, nullptr, 0, k2v, NC, N, NC);

    // lam_full
    k_lambda<<<1, 32, 0, stream>>>(lq1, lk1, lq2, lk2, lam);

    // apre = q1 @ k1v - lam_full * (q2 @ k2v)
    gemm_k<0, false><<<GB, blk, 0, stream>>>(q1, k1v, nullptr, nullptr, 1.f,  nullptr, nullptr, nullptr, 0, apre, N, NC, NC);
    gemm_k<0, false><<<GB, blk, 0, stream>>>(q2, k2v, nullptr, lam,    -1.f,  nullptr, nullptr, nullptr, 1, apre, N, NC, NC);

    // a = LN(apre) * (1 - lambda_init)   [in place]
    k_layernorm<<<LNG, 256, 0, stream>>>(apre, att_g, att_b, apre, N, 0.8f);

    // mh = (a @ out_w + out_b) + h + h_enc
    gemm_k<0, false><<<GB, blk, 0, stream>>>(apre, out_w, out_b, nullptr, 1.f, nullptr, h, henc, 0, mh, N, NC, NC);

    // f = LN(mh); out = mh + gelu(f @ ffn1 + b1) @ ffn2 + b2
    k_layernorm<<<LNG, 256, 0, stream>>>(mh, ffn_g, ffn_b, f, N, 1.0f);
    gemm_k<2, false><<<GB, blk, 0, stream>>>(f,  f1_w, f1_b, nullptr, 1.f, nullptr, nullptr, nullptr, 0, g1,  N, NC, NC);
    gemm_k<0, false><<<GB, blk, 0, stream>>>(g1, f2_w, f2_b, nullptr, 1.f, nullptr, mh,      nullptr, 0, out, N, NC, NC);
}
